// SlotAttentionTransformer_39900246179905
// MI455X (gfx1250) — matrix operand order repaired and checked by reading, not compiled, not run on hardware
//
#include <hip/hip_runtime.h>
#include <hip/hip_bf16.h>

// ---------------------------------------------------------------------------
// SlotAttentionTransformer for MI455X (gfx1250, wave32, WMMA bf16)
// Heavy 64x64 GEMMs (feature projections + k/v encoder MLPs) run on
// v_wmma_f32_16x16x32_bf16; weights staged to LDS with
// global_load_async_to_lds_b128 (ASYNCcnt); outputs written as b128 stores.
// ---------------------------------------------------------------------------

typedef __attribute__((ext_vector_type(16))) __bf16 v16bf;
typedef __attribute__((ext_vector_type(8)))  __bf16 v8bf;
typedef __attribute__((ext_vector_type(8)))  float  v8f;

#define B_   2
#define H_   128
#define W_   128
#define N_   (H_*W_)
#define K_   8
#define DIM  64
#define CAM_ 5
#define COL  8
#define DINO 768
#define SCALE_ 0.125f      // 64^-0.5
#define EPS_   1e-8f
#define MOM_   0.5f

// ------------------------------- helpers -----------------------------------

__device__ inline float wred16(float v) {         // reduce over one 16-lane half-group
  v += __shfl_xor(v, 1, 32); v += __shfl_xor(v, 2, 32);
  v += __shfl_xor(v, 4, 32); v += __shfl_xor(v, 8, 32);
  return v;
}
__device__ inline float wred32(float v) {         // full wave32 reduce
  v += __shfl_xor(v, 1, 32);  v += __shfl_xor(v, 2, 32);
  v += __shfl_xor(v, 4, 32);  v += __shfl_xor(v, 8, 32);
  v += __shfl_xor(v, 16, 32);
  return v;
}

__device__ inline v8f wmma_bf16(v16bf a, v16bf b, v8f c) {
  return __builtin_amdgcn_wmma_f32_16x16x32_bf16(false, a, false, b, (short)0, c, false, false);
}

// Async DMA: 16 bytes global -> LDS, tracked by ASYNCcnt (CDNA5).
// LDS operand = low 32 bits of the generic pointer (local->flat casts place
// the LDS offset in the low word).
__device__ inline void async_copy_b128(const void* g, void* l) {
  unsigned loff = (unsigned)(size_t)l;
  asm volatile("global_load_async_to_lds_b128 %0, %1, off"
               :: "v"(loff), "v"(g) : "memory");
}
__device__ inline void async_wait0() {
  asm volatile("s_wait_asynccnt 0" ::: "memory");
}

// B-fragment from LDS weight stored TRANSPOSED: Wt[n*64 + k] (bf16).
// Lane holds column n = nt*16 + c, K = kstep*32 + half*8 + e (e < 8), K = kstep*32 + 16 + half*8 + (e-8) (e >= 8).
__device__ inline v16bf load_bt(const __bf16* Wt, int kstep, int nt, int half, int c) {
  const __bf16* p = Wt + ((nt * 16 + c) * DIM + kstep * 32 + half * 8);
  const v8bf* pv = (const v8bf*)p;
  v8bf lo = pv[0], hi = pv[2];
  v16bf r;
#pragma unroll
  for (int e = 0; e < 8; ++e) { r[e] = lo[e]; r[8 + e] = hi[e]; }
  return r;
}

__device__ inline float bred64(volatile float* red, int d, float v) {
  red[d] = v; __syncthreads();
  for (int off = 32; off > 0; off >>= 1) { if (d < off) red[d] += red[d + off]; __syncthreads(); }
  float r = red[0]; __syncthreads();
  return r;
}
__device__ inline float bred256(volatile float* red, int t, float v) {
  red[t] = v; __syncthreads();
  for (int off = 128; off > 0; off >>= 1) { if (t < off) red[t] += red[t + off]; __syncthreads(); }
  float r = red[0]; __syncthreads();
  return r;
}

__device__ inline float adaln64(float sv, int d, int b, const float* cam,
                                const float* g, const float* bb,
                                const float* w, const float* bias,
                                volatile float* red) {
  float mean = bred64(red, d, sv) * (1.f / 64.f);
  float dv = sv - mean;
  float var = bred64(red, d, dv * dv) * (1.f / 64.f);
  float hn = dv * rsqrtf(var + 1e-5f) * g[d] + bb[d];
  float sc = bias[d], sf = bias[64 + d];
#pragma unroll
  for (int c = 0; c < CAM_; ++c) {
    float cm = cam[b * CAM_ + c];
    float si = cm / (1.f + expf(-cm));      // silu
    sc += si * w[c * 128 + d];
    sf += si * w[c * 128 + 64 + d];
  }
  return hn * (1.f + sc) + sf;
}

// ------------------------------ prep kernels -------------------------------

// f32 (k,n) row-major weight -> bf16 TRANSPOSED (n,k) so blocks can async-DMA
// it straight into LDS in B-fragment-friendly order.
__global__ void cvt_bf16_t_kernel(const float* __restrict__ src, __bf16* __restrict__ dst) {
  int i = blockIdx.x * blockDim.x + threadIdx.x;
  if (i < DIM * DIM) {
    int kk = i >> 6, dd = i & 63;
    dst[dd * DIM + kk] = (__bf16)src[i];
  }
}

// packs encoder param block: [ln_g, ln_b, b1, b2, gwx, gwy, grid_b] (7*64 f32)
__global__ void par_kernel(const float* lng, const float* lnb,
                           const float* b1, const float* b2,
                           const float* gw, const float* gb, float* par) {
  int d = threadIdx.x;
  par[d] = lng[d];        par[64 + d] = lnb[d];
  par[128 + d] = b1[d];   par[192 + d] = b2[d];
  par[256 + d] = gw[d] - gw[128 + d];          // grid_w[0]-grid_w[2]
  par[320 + d] = gw[64 + d] - gw[192 + d];     // grid_w[1]-grid_w[3]
  par[384 + d] = gb[d];
}

__global__ void grid_kernel(float* grid) {
  int n = blockIdx.x * 256 + threadIdx.x;
  if (n < N_) {
    int x = n % W_, y = n / W_;
    grid[2 * n]     = (2.f * x + 1.f) / (float)W_ - 1.f;
    grid[2 * n + 1] = (2.f * y + 1.f) / (float)H_ - 1.f;
  }
}

__global__ void bcast_kernel(const float* src, float* dst, int per, int reps) {
  int i = blockIdx.x * blockDim.x + threadIdx.x;
  if (i < per * reps) dst[i] = src[i % per];
}

__global__ void zero_kernel(float* p, int n) {
  int i = blockIdx.x * blockDim.x + threadIdx.x;
  if (i < n) p[i] = 0.f;
}

// --------------------- WMMA: LayerNorm + 64x64 GEMM ------------------------
// rows = B*N, 4 waves/block, 16 rows/wave.  out = LN(in) @ W  (f32 out)

__global__ __launch_bounds__(128)
void ln_gemm_kernel(const float* __restrict__ in, const float* __restrict__ g,
                    const float* __restrict__ bta, const __bf16* __restrict__ Wg,
                    float* __restrict__ out) {
  __shared__ __attribute__((aligned(16))) __bf16 sWt[DIM * DIM];
  __shared__ float sG[DIM], sB[DIM];
  __shared__ __attribute__((aligned(16))) float sT[4][16][DIM + 4];
  int tid = threadIdx.x;
  for (int i = tid; i < 512; i += 128)                       // 8KB async DMA
    async_copy_b128(Wg + (size_t)i * 8, &sWt[i * 8]);
  if (tid < DIM) { sG[tid] = g[tid]; sB[tid] = bta[tid]; }
  async_wait0();
  __syncthreads();

  int wave = tid >> 5, lane = tid & 31;
  int half = lane >> 4, m = lane & 15, c = m;
  size_t rowt = (size_t)blockIdx.x * 64 + wave * 16;
  size_t row = rowt + m;

  float xv[2][16];
  float s1 = 0.f;
  const float* prow = in + row * DIM;
#pragma unroll
  for (int s = 0; s < 2; ++s)
#pragma unroll
    for (int t = 0; t < 2; ++t)
#pragma unroll
      for (int i = 0; i < 8; ++i) {
        int dim = s * 32 + t * 16 + half * 8 + i;
        float v = prow[dim];
        xv[s][t * 8 + i] = v; s1 += v;
      }
  s1 += __shfl_xor(s1, 16, 32);
  float mean = s1 * (1.f / 64.f);
  float s2 = 0.f;
#pragma unroll
  for (int s = 0; s < 2; ++s)
#pragma unroll
    for (int e = 0; e < 16; ++e) { float d = xv[s][e] - mean; s2 += d * d; }
  s2 += __shfl_xor(s2, 16, 32);
  float rs = rsqrtf(s2 * (1.f / 64.f) + 1e-5f);

  v16bf a0, a1;
#pragma unroll
  for (int s = 0; s < 2; ++s)
#pragma unroll
    for (int t = 0; t < 2; ++t)
#pragma unroll
      for (int i = 0; i < 8; ++i) {
        int dim = s * 32 + t * 16 + half * 8 + i;
        float v = (xv[s][t * 8 + i] - mean) * rs * sG[dim] + sB[dim];
        if (s == 0) a0[t * 8 + i] = (__bf16)v; else a1[t * 8 + i] = (__bf16)v;
      }

  v8f zero = {};
  v8f acc[4] = {zero, zero, zero, zero};
#pragma unroll
  for (int nt = 0; nt < 4; ++nt) {
    acc[nt] = wmma_bf16(a0, load_bt(sWt, 0, nt, half, c), acc[nt]);
    acc[nt] = wmma_bf16(a1, load_bt(sWt, 1, nt, half, c), acc[nt]);
  }
  // stage D tile in LDS, then contiguous b128 row stores (32 f32 per lane)
#pragma unroll
  for (int nt = 0; nt < 4; ++nt)
#pragma unroll
    for (int r = 0; r < 8; ++r)
      sT[wave][r + 8 * half][nt * 16 + c] = acc[nt][r];
  float* dst = out + (rowt + m) * DIM + half * 32;
  const float* srcp = &sT[wave][m][half * 32];
#pragma unroll
  for (int j = 0; j < 8; ++j)
    ((float4*)dst)[j] = ((const float4*)srcp)[j];
}

// ---------------- WMMA: slot k/v encoder (LN -> GEMM -> relu -> GEMM) ------
// x = proj[b,n] + grid_embed(grid[n]-pos[b,k]); then mlp_enc.
// mode 0: store full output (B,KS,N,64); mode 1: logits = q . out * SCALE.

__global__ __launch_bounds__(128)
void enc_kernel(const float* __restrict__ proj, const float* __restrict__ pos,
                const float* __restrict__ grid,
                const __bf16* __restrict__ W1g, const __bf16* __restrict__ W2g,
                const float* __restrict__ par, const float* __restrict__ q,
                float* __restrict__ out, int KS, int mode) {
  __shared__ __attribute__((aligned(16))) __bf16 sW1t[DIM * DIM];
  __shared__ __attribute__((aligned(16))) __bf16 sW2t[DIM * DIM];
  __shared__ float sPar[7 * DIM];
  __shared__ __attribute__((aligned(16))) float sT[4][16][DIM + 4];
  int tid = threadIdx.x;
  for (int i = tid; i < 512; i += 128) {                     // 2 x 8KB async DMA
    async_copy_b128(W1g + (size_t)i * 8, &sW1t[i * 8]);
    async_copy_b128(W2g + (size_t)i * 8, &sW2t[i * 8]);
  }
  for (int i = tid; i < 7 * DIM; i += 128) sPar[i] = par[i];
  async_wait0();
  __syncthreads();
  const float* lng = sPar;        const float* lnb = sPar + 64;
  const float* bb1 = sPar + 128;  const float* bb2 = sPar + 192;
  const float* gwx = sPar + 256;  const float* gwy = sPar + 320;
  const float* gbb = sPar + 384;

  int wave = tid >> 5, lane = tid & 31;
  int half = lane >> 4, m = lane & 15, c = m;
  int b = blockIdx.z, k = blockIdx.y;
  size_t rowt = (size_t)blockIdx.x * 64 + wave * 16;
  size_t n = rowt + m;

  float px = 0.f, py = 0.f;
  if (pos) { px = pos[(b * KS + k) * 2]; py = pos[(b * KS + k) * 2 + 1]; }
  float rx = grid[2 * n] - px, ry = grid[2 * n + 1] - py;
  const float* prow = proj + ((size_t)b * N_ + n) * DIM;

  // x = proj + ge; row LayerNorm split across lane pair (L, L+16)
  float xv[2][16];
  float s1 = 0.f;
#pragma unroll
  for (int s = 0; s < 2; ++s)
#pragma unroll
    for (int t = 0; t < 2; ++t)
#pragma unroll
      for (int i = 0; i < 8; ++i) {
        int dim = s * 32 + t * 16 + half * 8 + i;
        float v = prow[dim] + rx * gwx[dim] + ry * gwy[dim] + gbb[dim];
        xv[s][t * 8 + i] = v; s1 += v;
      }
  s1 += __shfl_xor(s1, 16, 32);
  float mean = s1 * (1.f / 64.f);
  float s2 = 0.f;
#pragma unroll
  for (int s = 0; s < 2; ++s)
#pragma unroll
    for (int e = 0; e < 16; ++e) { float d = xv[s][e] - mean; s2 += d * d; }
  s2 += __shfl_xor(s2, 16, 32);
  float rs = rsqrtf(s2 * (1.f / 64.f) + 1e-5f);

  v16bf a0, a1;
#pragma unroll
  for (int s = 0; s < 2; ++s)
#pragma unroll
    for (int t = 0; t < 2; ++t)
#pragma unroll
      for (int i = 0; i < 8; ++i) {
        int dim = s * 32 + t * 16 + half * 8 + i;
        float v = (xv[s][t * 8 + i] - mean) * rs * lng[dim] + lnb[dim];
        if (s == 0) a0[t * 8 + i] = (__bf16)v; else a1[t * 8 + i] = (__bf16)v;
      }

  v8f zero = {};
  v8f acc[4] = {zero, zero, zero, zero};
#pragma unroll
  for (int nt = 0; nt < 4; ++nt) {
    acc[nt] = wmma_bf16(a0, load_bt(sW1t, 0, nt, half, c), acc[nt]);
    acc[nt] = wmma_bf16(a1, load_bt(sW1t, 1, nt, half, c), acc[nt]);
  }
  // bias + relu, transpose through wave-private LDS tile (D layout -> A layout)
#pragma unroll
  for (int nt = 0; nt < 4; ++nt)
#pragma unroll
    for (int r = 0; r < 8; ++r)
      sT[wave][r + 8 * half][nt * 16 + c] = fmaxf(acc[nt][r] + bb1[nt * 16 + c], 0.f);

  v16bf a20, a21;
#pragma unroll
  for (int s = 0; s < 2; ++s)
#pragma unroll
    for (int t = 0; t < 2; ++t)
#pragma unroll
      for (int i = 0; i < 8; ++i) {
        int dim = s * 32 + t * 16 + half * 8 + i;
        float v = sT[wave][m][dim];
        if (s == 0) a20[t * 8 + i] = (__bf16)v; else a21[t * 8 + i] = (__bf16)v;
      }
  v8f acc2[4] = {zero, zero, zero, zero};
#pragma unroll
  for (int nt = 0; nt < 4; ++nt) {
    acc2[nt] = wmma_bf16(a20, load_bt(sW2t, 0, nt, half, c), acc2[nt]);
    acc2[nt] = wmma_bf16(a21, load_bt(sW2t, 1, nt, half, c), acc2[nt]);
  }

  if (mode == 0) {
    // stage D+bias in LDS, then contiguous b128 row stores
#pragma unroll
    for (int nt = 0; nt < 4; ++nt)
#pragma unroll
      for (int r = 0; r < 8; ++r)
        sT[wave][r + 8 * half][nt * 16 + c] = acc2[nt][r] + bb2[nt * 16 + c];
    float* dst = out + ((size_t)(b * KS + k) * N_ + rowt + m) * DIM + half * 32;
    const float* srcp = &sT[wave][m][half * 32];
#pragma unroll
    for (int j = 0; j < 8; ++j)
      ((float4*)dst)[j] = ((const float4*)srcp)[j];
  } else {
    const float* qp = q + (b * KS + k) * DIM;
    float pr[8];
#pragma unroll
    for (int r = 0; r < 8; ++r) pr[r] = 0.f;
#pragma unroll
    for (int nt = 0; nt < 4; ++nt) {
      float qq = qp[nt * 16 + c];
#pragma unroll
      for (int r = 0; r < 8; ++r) pr[r] += (acc2[nt][r] + bb2[nt * 16 + c]) * qq;
    }
#pragma unroll
    for (int r = 0; r < 8; ++r) pr[r] = wred16(pr[r]);
    if (c == 0) {
      float* lb = out + (size_t)(b * KS + k) * N_ + rowt;
#pragma unroll
      for (int r = 0; r < 8; ++r) lb[r + 8 * half] = pr[r] * SCALE_;
    }
  }
}

// --------------------------- small VALU kernels ----------------------------

__global__ void lncolor_kernel(const float* __restrict__ x, const float* __restrict__ g,
                               const float* __restrict__ bb, float* __restrict__ y) {
  int idx = blockIdx.x * 256 + threadIdx.x;
  if (idx >= B_ * N_) return;
  const float* xr = x + (size_t)idx * COL;
  float v[COL]; float s = 0.f;
#pragma unroll
  for (int j = 0; j < COL; ++j) { v[j] = xr[j]; s += v[j]; }
  float m = s * (1.f / COL);
  float s2 = 0.f;
#pragma unroll
  for (int j = 0; j < COL; ++j) { float d = v[j] - m; s2 += d * d; }
  float rs = rsqrtf(s2 * (1.f / COL) + 1e-5f);
  float* yr = y + (size_t)idx * COL;
#pragma unroll
  for (int j = 0; j < COL; ++j) yr[j] = (v[j] - m) * rs * g[j] + bb[j];
}

__global__ void q_kernel(const float* __restrict__ slot, const float* __restrict__ cam,
                         const float* g, const float* bb, const float* w, const float* bias,
                         const float* __restrict__ wq, float* __restrict__ qout, int KS) {
  int b = blockIdx.x / KS, k = blockIdx.x % KS;
  int d = threadIdx.x;
  __shared__ float red[64];
  __shared__ float sh[64];
  float sv = slot[((size_t)b * KS + k) * DIM + d];
  float x = adaln64(sv, d, b, cam, g, bb, w, bias, red);
  sh[d] = x; __syncthreads();
  float o = 0.f;
#pragma unroll 8
  for (int j = 0; j < DIM; ++j) o += sh[j] * wq[j * DIM + d];
  qout[((size_t)b * KS + k) * DIM + d] = o;
}

__global__ void slot_mlp_kernel(float* __restrict__ slot, const float* __restrict__ cam,
                                const float* g, const float* bb, const float* w, const float* bias,
                                const float* w1, const float* b1,
                                const float* w2, const float* b2, int KS) {
  int b = blockIdx.x / KS, k = blockIdx.x % KS;
  int d = threadIdx.x;
  __shared__ float red[64];
  __shared__ float sh[64];
  __shared__ float h1[64];
  float* sp = slot + ((size_t)b * KS + k) * DIM;
  float sv = sp[d];
  float x = adaln64(sv, d, b, cam, g, bb, w, bias, red);
  sh[d] = x; __syncthreads();
  float a1 = b1[d];
#pragma unroll 8
  for (int j = 0; j < DIM; ++j) a1 += sh[j] * w1[j * DIM + d];
  float ge = 0.5f * a1 * (1.f + erff(a1 * 0.70710678f));
  h1[d] = ge; __syncthreads();
  float o = b2[d];
#pragma unroll 8
  for (int j = 0; j < DIM; ++j) o += h1[j] * w2[j * DIM + d];
  sp[d] = sv + o;
}

__global__ void dotq_kernel(const float* __restrict__ q, const float* __restrict__ kbg,
                            float* __restrict__ out) {
  int idx = blockIdx.x * 256 + threadIdx.x;
  if (idx >= B_ * N_) return;
  int b = idx / N_, n = idx % N_;
  const float* kp = kbg + ((size_t)b * N_ + n) * DIM;
  const float* qp = q + b * DIM;
  float s = 0.f;
#pragma unroll 8
  for (int j = 0; j < DIM; ++j) s += qp[j] * kp[j];
  out[(size_t)b * N_ + n] = s * SCALE_;
}

__global__ void attn_kernel(const float* __restrict__ lbg, const float* __restrict__ lfg,
                            float* __restrict__ attn, float* __restrict__ sums) {
  int idx = blockIdx.x * 256 + threadIdx.x;
  int b = idx / N_, n = idx % N_;
  float l[K_];
  l[0] = lbg[(size_t)b * N_ + n];
#pragma unroll
  for (int k = 1; k < K_; ++k) l[k] = lfg[((size_t)b * 7 + (k - 1)) * N_ + n];
  float mx = l[0];
#pragma unroll
  for (int k = 1; k < K_; ++k) mx = fmaxf(mx, l[k]);
  float s = 0.f;
#pragma unroll
  for (int k = 0; k < K_; ++k) { l[k] = expf(l[k] - mx); s += l[k]; }
  float inv = 1.f / s;
  int lane = threadIdx.x & 31;
#pragma unroll
  for (int k = 0; k < K_; ++k) {
    float a = l[k] * inv + EPS_;
    attn[((size_t)b * K_ + k) * N_ + n] = a;
    float r = wred32(a);
    if (lane == 0) atomicAdd(&sums[b * K_ + k], r);
  }
}

__global__ void pos_kernel(const float* __restrict__ attn, const float* __restrict__ grid,
                           const float* __restrict__ posw, const float* __restrict__ posb,
                           const float* __restrict__ sums, float* __restrict__ pos, int last) {
  int k = blockIdx.x, b = blockIdx.y;
  const float* at = attn + ((size_t)b * K_ + 1 + k) * N_;
  float ax = 0.f, ay = 0.f, px = 0.f, py = 0.f;
  for (int n = threadIdx.x; n < N_; n += 256) {
    float a = at[n];
    ax += a * grid[2 * n]; ay += a * grid[2 * n + 1];
    if (last) { px += a * posw[2 * n]; py += a * posw[2 * n + 1]; }
  }
  ax = wred32(ax); ay = wred32(ay); px = wred32(px); py = wred32(py);
  __shared__ float sm[4][8];
  int wv = threadIdx.x >> 5, ln = threadIdx.x & 31;
  if (ln == 0) { sm[0][wv] = ax; sm[1][wv] = ay; sm[2][wv] = px; sm[3][wv] = py; }
  __syncthreads();
  if (threadIdx.x == 0) {
    float AX = 0, AY = 0, PX = 0, PY = 0;
    for (int w2 = 0; w2 < 8; ++w2) { AX += sm[0][w2]; AY += sm[1][w2]; PX += sm[2][w2]; PY += sm[3][w2]; }
    float inv = 1.f / sums[b * K_ + 1 + k];
    int pi = (b * 7 + k) * 2;
    float nx = AX * inv * (1.f - MOM_) + pos[pi] * MOM_;
    float ny = AY * inv * (1.f - MOM_) + pos[pi + 1] * MOM_;
    if (last) {
      nx += tanhf(PX * inv + posb[0]) * 0.1f;
      ny += tanhf(PY * inv + posb[1]) * 0.1f;
    }
    pos[pi] = nx; pos[pi + 1] = ny;
  }
}

__global__ void slot_update_kernel(const float* __restrict__ attn, const float* __restrict__ vfg,
                                   const float* __restrict__ vbg, const float* __restrict__ sums,
                                   float* __restrict__ slot_fg, float* __restrict__ slot_bg) {
  int k = blockIdx.x, b = blockIdx.y;
  int d = threadIdx.x & 63, ch = threadIdx.x >> 6;
  const float* at = attn + ((size_t)b * K_ + k) * N_;
  const float* v = (k == 0) ? (vbg + (size_t)b * N_ * DIM)
                            : (vfg + ((size_t)b * 7 + (k - 1)) * N_ * DIM);
  float acc = 0.f;
  for (int n = ch; n < N_; n += 4) acc += at[n] * v[(size_t)n * DIM + d];
  __shared__ float sm[4][DIM];
  sm[ch][d] = acc; __syncthreads();
  if (ch == 0) {
    float t = (sm[0][d] + sm[1][d] + sm[2][d] + sm[3][d]) / sums[b * K_ + k];
    if (k == 0) slot_bg[(size_t)b * DIM + d] += t;
    else        slot_fg[((size_t)b * 7 + (k - 1)) * DIM + d] += t;
  }
}

__global__ void color_kernel(const float* __restrict__ attn, const float* __restrict__ fc,
                             const float* __restrict__ sums,
                             float* __restrict__ cfg, float* __restrict__ cbg) {
  int k = blockIdx.x, b = blockIdx.y;
  int d = threadIdx.x & 7, ch = threadIdx.x >> 3;   // 32 chunks
  const float* at = attn + ((size_t)b * K_ + k) * N_;
  float acc = 0.f;
  for (int n = ch; n < N_; n += 32) acc += at[n] * fc[((size_t)b * N_ + n) * COL + d];
  __shared__ float sm[32][COL];
  sm[ch][d] = acc; __syncthreads();
  if (ch == 0) {
    float t = 0.f;
    for (int i = 0; i < 32; ++i) t += sm[i][d];
    t /= sums[b * K_ + k];
    if (k == 0) cbg[b * COL + d] = t;
    else        cfg[((size_t)b * 7 + (k - 1)) * COL + d] = t;
  }
}

__global__ void gen_kernel(const float* __restrict__ slot_fg, const float* __restrict__ cfg,
                           const float* __restrict__ gw, const float* __restrict__ gb,
                           const float* __restrict__ lg, const float* __restrict__ lb,
                           float* __restrict__ out) {
  int b = blockIdx.x / 7, k = blockIdx.x % 7;
  __shared__ float s[72];
  __shared__ float red[256];
  int t = threadIdx.x;
  if (t < 64)      s[t] = slot_fg[((size_t)b * 7 + k) * DIM + t];
  else if (t < 72) s[t] = cfg[((size_t)b * 7 + k) * COL + (t - 64)];
  __syncthreads();
  float loc[3];
  float psum = 0.f;
#pragma unroll
  for (int i = 0; i < 3; ++i) {
    int d = t + i * 256;
    float a = gb[d];
    for (int j = 0; j < 72; ++j) a += s[j] * gw[j * DINO + d];
    loc[i] = a; psum += a;
  }
  float mean = bred256(red, t, psum) * (1.f / DINO);
  float pv = 0.f;
#pragma unroll
  for (int i = 0; i < 3; ++i) { float dd = loc[i] - mean; pv += dd * dd; }
  float var = bred256(red, t, pv) * (1.f / DINO);
  float rs = rsqrtf(var + 1e-5f);
#pragma unroll
  for (int i = 0; i < 3; ++i) {
    int d = t + i * 256;
    out[((size_t)b * 7 + k) * DINO + d] = (loc[i] - mean) * rs * lg[d] + lb[d];
  }
}

__global__ void pack_kernel(const float* __restrict__ slot_bg, const float* __restrict__ cbg,
                            const float* __restrict__ pos,
                            float* out0, float* out2, float* out3) {
  int t = threadIdx.x;
  if (t < B_ * 72) {
    int b = t / 72, j = t % 72;
    out0[t] = (j < 64) ? slot_bg[b * DIM + j] : cbg[b * COL + (j - 64)];
  }
  if (t < B_ * 7 * 2) out2[t] = pos[t];
  if (t < B_ * 7)     out3[t] = 1.0f;
}

// ------------------------------- host side ---------------------------------

enum {
  I_FEAT = 0, I_FCOLOR, I_CAM, I_FGPOS, I_SLOTSFG, I_SLOTSBG,
  I_POSW, I_POSB, I_GRIDW, I_GRIDB,
  I_WKFG, I_WVFG, I_WKBG, I_WVBG,
  I_EFG_LNG, I_EFG_LNB, I_EFG_W1, I_EFG_B1, I_EFG_W2, I_EFG_B2,
  I_EBG_LNG, I_EBG_LNB, I_EBG_W1, I_EBG_B1, I_EBG_W2, I_EBG_B2,
  I_AQF_G, I_AQF_B, I_AQF_W, I_AQF_BIAS,
  I_AQB_G, I_AQB_B, I_AQB_W, I_AQB_BIAS,
  I_AMF_G, I_AMF_B, I_AMF_W, I_AMF_BIAS,
  I_AMB_G, I_AMB_B, I_AMB_W, I_AMB_BIAS,
  I_WQFG, I_WQBG, I_NFG, I_NFB, I_NCG, I_NCB,
  I_MF_W1, I_MF_B1, I_MF_W2, I_MF_B2,
  I_MB_W1, I_MB_B1, I_MB_W2, I_MB_B2,
  I_GENW, I_GENB, I_GLNG, I_GLNB
};

#define FIN(i) ((const float*)d_in[(i)])

extern "C" void kernel_launch(void* const* d_in, const int* in_sizes, int n_in,
                              void* d_out, int out_size, void* d_ws, size_t ws_size,
                              hipStream_t stream) {
  (void)in_sizes; (void)n_in; (void)out_size; (void)ws_size;

  float* W = (float*)d_ws;
  size_t o = 0;
  auto alloc = [&](size_t n) { float* p = W + o; o += n; return p; };
  const size_t PN = (size_t)B_ * N_ * DIM;

  float* kproj_fg = alloc(PN);
  float* vproj_fg = alloc(PN);
  float* kproj_bg = alloc(PN);
  float* vproj_bg = alloc(PN);
  float* k_bg     = alloc(PN);
  float* v_bg     = alloc(PN);
  float* v_fg     = alloc((size_t)B_ * 7 * N_ * DIM);
  float* logits_fg = alloc((size_t)B_ * 7 * N_);
  float* logits_bg = alloc((size_t)B_ * N_);
  float* q_fg   = alloc(B_ * 7 * DIM);
  float* q_bg   = alloc(B_ * DIM);
  float* slot_fg = alloc(B_ * 7 * DIM);
  float* slot_bg = alloc(B_ * DIM);
  float* posbuf = alloc(B_ * 7 * 2);
  float* sums   = alloc(B_ * K_);
  float* fc     = alloc((size_t)B_ * N_ * COL);
  float* cfg    = alloc(B_ * 7 * COL);
  float* cbg    = alloc(B_ * COL);
  float* gridb  = alloc(N_ * 2);
  float* parfg  = alloc(7 * DIM);
  float* parbg  = alloc(7 * DIM);
  __bf16* wbf   = (__bf16*)alloc(8 * DIM * DIM / 2);   // 16B-aligned (all counts %4==0)

  // ---- prep: transposed bf16 weights, fused grid-embed params, broadcasts ----
  const float* wsrc[8] = {FIN(I_WKFG), FIN(I_WVFG), FIN(I_WKBG), FIN(I_WVBG),
                          FIN(I_EFG_W1), FIN(I_EFG_W2), FIN(I_EBG_W1), FIN(I_EBG_W2)};
  for (int i = 0; i < 8; ++i)
    cvt_bf16_t_kernel<<<16, 256, 0, stream>>>(wsrc[i], wbf + (size_t)i * DIM * DIM);
  par_kernel<<<1, 64, 0, stream>>>(FIN(I_EFG_LNG), FIN(I_EFG_LNB), FIN(I_EFG_B1),
                                   FIN(I_EFG_B2), FIN(I_GRIDW), FIN(I_GRIDB), parfg);
  par_kernel<<<1, 64, 0, stream>>>(FIN(I_EBG_LNG), FIN(I_EBG_LNB), FIN(I_EBG_B1),
                                   FIN(I_EBG_B2), FIN(I_GRIDW), FIN(I_GRIDB), parbg);
  grid_kernel<<<N_ / 256, 256, 0, stream>>>(gridb);
  bcast_kernel<<<(B_ * 7 * DIM + 255) / 256, 256, 0, stream>>>(FIN(I_SLOTSFG), slot_fg, 7 * DIM, B_);
  bcast_kernel<<<1, 256, 0, stream>>>(FIN(I_SLOTSBG), slot_bg, DIM, B_);
  bcast_kernel<<<1, 64, 0, stream>>>(FIN(I_FGPOS), posbuf, 14, B_);

  // ---- projections: LN(feat) @ {wk_fg, wv_fg, wk_bg, wv_bg} (WMMA) ----
  int gP = (B_ * N_) / 64;
  ln_gemm_kernel<<<gP, 128, 0, stream>>>(FIN(I_FEAT), FIN(I_NFG), FIN(I_NFB), wbf + 0 * 4096, kproj_fg);
  ln_gemm_kernel<<<gP, 128, 0, stream>>>(FIN(I_FEAT), FIN(I_NFG), FIN(I_NFB), wbf + 1 * 4096, vproj_fg);
  ln_gemm_kernel<<<gP, 128, 0, stream>>>(FIN(I_FEAT), FIN(I_NFG), FIN(I_NFB), wbf + 2 * 4096, kproj_bg);
  ln_gemm_kernel<<<gP, 128, 0, stream>>>(FIN(I_FEAT), FIN(I_NFG), FIN(I_NFB), wbf + 3 * 4096, vproj_bg);

  // ---- bg encoder (once): k_bg, v_bg (WMMA) ----
  dim3 gbg(N_ / 64, 1, B_);
  enc_kernel<<<gbg, 128, 0, stream>>>(kproj_bg, nullptr, gridb, wbf + 6 * 4096, wbf + 7 * 4096,
                                      parbg, nullptr, k_bg, 1, 0);
  enc_kernel<<<gbg, 128, 0, stream>>>(vproj_bg, nullptr, gridb, wbf + 6 * 4096, wbf + 7 * 4096,
                                      parbg, nullptr, v_bg, 1, 0);
  lncolor_kernel<<<(B_ * N_) / 256, 256, 0, stream>>>(FIN(I_FCOLOR), FIN(I_NCG), FIN(I_NCB), fc);

  float* attn_out = (float*)d_out + B_ * 72;   // out1 region (B,8,N)
  float* out0 = (float*)d_out;
  float* out2 = attn_out + (size_t)B_ * K_ * N_;
  float* out3 = out2 + B_ * 7 * 2;
  float* out4 = out3 + B_ * 7;

  for (int it = 0; it < 4; ++it) {
    int last = (it == 3);
    q_kernel<<<B_ * 7, 64, 0, stream>>>(slot_fg, FIN(I_CAM), FIN(I_AQF_G), FIN(I_AQF_B),
                                        FIN(I_AQF_W), FIN(I_AQF_BIAS), FIN(I_WQFG), q_fg, 7);
    q_kernel<<<B_, 64, 0, stream>>>(slot_bg, FIN(I_CAM), FIN(I_AQB_G), FIN(I_AQB_B),
                                    FIN(I_AQB_W), FIN(I_AQB_BIAS), FIN(I_WQBG), q_bg, 1);
    dim3 gfg(N_ / 64, 7, B_);
    enc_kernel<<<gfg, 128, 0, stream>>>(kproj_fg, posbuf, gridb, wbf + 4 * 4096, wbf + 5 * 4096,
                                        parfg, q_fg, logits_fg, 7, 1);
    enc_kernel<<<gfg, 128, 0, stream>>>(vproj_fg, posbuf, gridb, wbf + 4 * 4096, wbf + 5 * 4096,
                                        parfg, nullptr, v_fg, 7, 0);
    dotq_kernel<<<(B_ * N_) / 256, 256, 0, stream>>>(q_bg, k_bg, logits_bg);
    zero_kernel<<<1, 32, 0, stream>>>(sums, B_ * K_);
    attn_kernel<<<(B_ * N_) / 256, 256, 0, stream>>>(logits_bg, logits_fg, attn_out, sums);
    pos_kernel<<<dim3(7, B_), 256, 0, stream>>>(attn_out, gridb, FIN(I_POSW), FIN(I_POSB),
                                                sums, posbuf, last);
    if (!last) {
      slot_update_kernel<<<dim3(K_, B_), 256, 0, stream>>>(attn_out, v_fg, v_bg, sums,
                                                           slot_fg, slot_bg);
      slot_mlp_kernel<<<B_, 64, 0, stream>>>(slot_bg, FIN(I_CAM), FIN(I_AMB_G), FIN(I_AMB_B),
                                             FIN(I_AMB_W), FIN(I_AMB_BIAS),
                                             FIN(I_MB_W1), FIN(I_MB_B1), FIN(I_MB_W2), FIN(I_MB_B2), 1);
      slot_mlp_kernel<<<B_ * 7, 64, 0, stream>>>(slot_fg, FIN(I_CAM), FIN(I_AMF_G), FIN(I_AMF_B),
                                                 FIN(I_AMF_W), FIN(I_AMF_BIAS),
                                                 FIN(I_MF_W1), FIN(I_MF_B1), FIN(I_MF_W2), FIN(I_MF_B2), 7);
    } else {
      color_kernel<<<dim3(K_, B_), 256, 0, stream>>>(attn_out, fc, sums, cfg, cbg);
    }
  }

  gen_kernel<<<B_ * 7, 256, 0, stream>>>(slot_fg, cfg, FIN(I_GENW), FIN(I_GENB),
                                         FIN(I_GLNG), FIN(I_GLNB), out4);
  pack_kernel<<<1, 256, 0, stream>>>(slot_bg, cbg, posbuf, out0, out2, out3);
}
